// HeteroRelEdgeConvLayer_82308753260705
// MI455X (gfx1250) — matrix operand order repaired and checked by reading, not compiled, not run on hardware
//
#include <hip/hip_runtime.h>
#include <hip/hip_bf16.h>

typedef __attribute__((ext_vector_type(16))) __bf16 v16bf;
typedef __attribute__((ext_vector_type(8)))  float  v8f;

#define N_NODES   50000
#define N_EDGES   800000
#define D_IN      96
#define D_EDGE    32
#define D_FEAT    128     // D_IN + D_EDGE
#define D_OUT     96
#define N_TYPES   4
#define BINS      (N_NODES * N_TYPES)   // 200000 sort bins (key = dst*4 + type)

#define TILE_EDGES   16
#define N_TILES_OUT  6    // D_OUT / 16
#define N_KCHUNK     4    // D_FEAT / 32
#define BLOCK_THREADS 256
#define WAVES_PER_BLOCK (BLOCK_THREADS / 32)

#define FRAG_ELEMS 512                                  // 32 lanes * 16 bf16
#define N_FRAGS    (N_TYPES * N_KCHUNK * N_TILES_OUT)   // 96
#define SW_ELEMS   (N_FRAGS * FRAG_ELEMS)               // 49152 bf16 = 96 KB
#define SMEM_BYTES (SW_ELEMS * 2 + N_TYPES * D_OUT * 4) // + bias fp32

#define NSB ((BINS + 255) / 256)                        // 782 scan blocks

// ---------------------------------------------------------------------------
// Small utility kernels: output zeroing + counting sort of edges by (dst,type)
// ---------------------------------------------------------------------------
__global__ void zero_out_kernel(float* __restrict__ out, int n) {
    int i = blockIdx.x * blockDim.x + threadIdx.x;
    if (i < n) out[i] = 0.0f;
}

__global__ void zero_bins_kernel(int* __restrict__ bins, int n) {
    int i = blockIdx.x * blockDim.x + threadIdx.x;
    if (i < n) bins[i] = 0;
}

__global__ void hist_kernel(const int* __restrict__ ei, const int* __restrict__ et,
                            int* __restrict__ bins) {
    int e = blockIdx.x * blockDim.x + threadIdx.x;
    if (e < N_EDGES) {
        int key = ei[N_EDGES + e] * N_TYPES + et[e];   // dst-major
        atomicAdd(&bins[key], 1);
    }
}

__global__ void scan_bins_kernel(int* __restrict__ bins, int* __restrict__ bsum) {
    __shared__ int s[256];
    int t = threadIdx.x;
    int i = blockIdx.x * 256 + t;
    int v = (i < BINS) ? bins[i] : 0;
    s[t] = v;
    __syncthreads();
    for (int off = 1; off < 256; off <<= 1) {
        int x = 0;
        if (t >= off) x = s[t - off];
        __syncthreads();
        s[t] += x;
        __syncthreads();
    }
    int incl = s[t];
    if (i < BINS) bins[i] = incl - v;          // exclusive within block
    if (t == 255) bsum[blockIdx.x] = incl;     // block total
}

__global__ void scan_sums_kernel(int* __restrict__ bsum, int nb) {
    __shared__ int s[1024];
    int t = threadIdx.x;
    int v = (t < nb) ? bsum[t] : 0;
    s[t] = v;
    __syncthreads();
    for (int off = 1; off < 1024; off <<= 1) {
        int x = 0;
        if (t >= off) x = s[t - off];
        __syncthreads();
        s[t] += x;
        __syncthreads();
    }
    if (t < nb) bsum[t] = s[t] - v;            // exclusive
}

__global__ void add_offsets_kernel(int* __restrict__ bins, const int* __restrict__ bsum) {
    int i = blockIdx.x * blockDim.x + threadIdx.x;
    if (i < BINS) bins[i] += bsum[i >> 8];
}

__global__ void scatter_kernel(const int* __restrict__ ei, const int* __restrict__ et,
                               int* __restrict__ bins, int* __restrict__ perm) {
    int e = blockIdx.x * blockDim.x + threadIdx.x;
    if (e < N_EDGES) {
        int key = ei[N_EDGES + e] * N_TYPES + et[e];
        int pos = atomicAdd(&bins[key], 1);
        perm[pos] = e;
    }
}

// ---------------------------------------------------------------------------
// Main fused WMMA kernel over dst-sorted edges
// ---------------------------------------------------------------------------
template <int BASE>
__device__ __forceinline__ void pack8(v16bf& v, float4 a, float4 b) {
    v[BASE + 0] = (__bf16)a.x; v[BASE + 1] = (__bf16)a.y;
    v[BASE + 2] = (__bf16)a.z; v[BASE + 3] = (__bf16)a.w;
    v[BASE + 4] = (__bf16)b.x; v[BASE + 5] = (__bf16)b.y;
    v[BASE + 6] = (__bf16)b.z; v[BASE + 7] = (__bf16)b.w;
}

__global__ __launch_bounds__(BLOCK_THREADS)
void rel_edge_conv_wmma(const float* __restrict__ x,
                        const float* __restrict__ ef,
                        const int*   __restrict__ ei,      // [2, E]
                        const int*   __restrict__ et,      // [E]
                        const float* __restrict__ W,       // [T, 128, 96]
                        const float* __restrict__ bia,     // [T, 96]
                        const int*   __restrict__ perm,    // dst-sorted edge ids
                        float*       __restrict__ out,     // [N, 96]
                        int nTiles, int totalWaves) {
    extern __shared__ __align__(32) char smem[];
    __bf16* sW    = (__bf16*)smem;                         // swizzled bf16 fragments
    float*  sBias = (float*)(smem + SW_ELEMS * 2);

    // ---- Stage weights into LDS, bf16, fragment-swizzled ----------------
    // Fragment f = (t*4 + c)*6 + n holds B tile W[t][c*32..+31][n*16..+15].
    // Lane l (col = l&15, khalf = l>>4) owns 16 contiguous bf16 at offset l*16:
//   element i -> W[t][c*32 + (i<8 ? khalf*8 + i : 16 + khalf*8 + i-8)][n*16 + col]
    for (int idx = threadIdx.x; idx < SW_ELEMS; idx += BLOCK_THREADS) {
        int f    = idx >> 9;
        int r    = idx & 511;
        int lane = r >> 4;
        int i    = r & 15;
        int t    = f / (N_KCHUNK * N_TILES_OUT);
        int rem  = f % (N_KCHUNK * N_TILES_OUT);
        int c    = rem / N_TILES_OUT;
        int n    = rem % N_TILES_OUT;
        int col  = lane & 15;
        int kh   = lane >> 4;
        int k    = c * 32 + (i < 8 ? kh * 8 + i : 16 + kh * 8 + (i - 8));
        sW[idx]  = (__bf16)W[(t * D_FEAT + k) * D_OUT + n * 16 + col];
    }
    for (int idx = threadIdx.x; idx < N_TYPES * D_OUT; idx += BLOCK_THREADS)
        sBias[idx] = bia[idx];
    __syncthreads();

    const int lane  = threadIdx.x & 31;
    const int gwave = blockIdx.x * WAVES_PER_BLOCK + (threadIdx.x >> 5);
    const int row   = lane & 15;   // edge row this lane feeds (A & meta)
    const int h     = lane >> 4;   // K-half selector for A layout
    const int col   = row;         // output column owned by this lane in C/D

    for (int tile = gwave; tile < nTiles; tile += totalWaves) {
        int e     = tile * TILE_EDGES + row;
        bool vld  = (e < N_EDGES);
        int  ec   = vld ? e : (N_EDGES - 1);
        int  pe   = perm[ec];                 // sorted edge id
        int  src  = ei[pe];
        int  dst  = ei[N_EDGES + pe];
        int  ety  = vld ? et[pe] : -1;

        // ---- Build A (16 edges x 128 feat) as 4 bf16 K-chunks -----------
        v16bf A[N_KCHUNK];
        {
            const float* xrow = x + (long)src * D_IN;
#pragma unroll
            for (int c = 0; c < 3; ++c) {
                float4 r1a = *(const float4*)(xrow + c * 32 + h * 8);
                float4 r1b = *(const float4*)(xrow + c * 32 + h * 8 + 4);
                float4 r2a = *(const float4*)(xrow + c * 32 + 16 + h * 8);
                float4 r2b = *(const float4*)(xrow + c * 32 + 16 + h * 8 + 4);
                pack8<0>(A[c], r1a, r1b);
                pack8<8>(A[c], r2a, r2b);
            }
            const float* es = ef + (long)src * D_EDGE;
            const float* ed = ef + (long)dst * D_EDGE;
            float4 s1a = *(const float4*)(es + h * 8);
            float4 s1b = *(const float4*)(es + h * 8 + 4);
            float4 s2a = *(const float4*)(es + 16 + h * 8);
            float4 s2b = *(const float4*)(es + 16 + h * 8 + 4);
            float4 d1a = *(const float4*)(ed + h * 8);
            float4 d1b = *(const float4*)(ed + h * 8 + 4);
            float4 d2a = *(const float4*)(ed + 16 + h * 8);
            float4 d2b = *(const float4*)(ed + 16 + h * 8 + 4);
            float4 a1a = {fabsf(s1a.x - d1a.x), fabsf(s1a.y - d1a.y),
                          fabsf(s1a.z - d1a.z), fabsf(s1a.w - d1a.w)};
            float4 a1b = {fabsf(s1b.x - d1b.x), fabsf(s1b.y - d1b.y),
                          fabsf(s1b.z - d1b.z), fabsf(s1b.w - d1b.w)};
            float4 a2a = {fabsf(s2a.x - d2a.x), fabsf(s2a.y - d2a.y),
                          fabsf(s2a.z - d2a.z), fabsf(s2a.w - d2a.w)};
            float4 a2b = {fabsf(s2b.x - d2b.x), fabsf(s2b.y - d2b.y),
                          fabsf(s2b.z - d2b.z), fabsf(s2b.w - d2b.w)};
            pack8<0>(A[3], a1a, a1b);
            pack8<8>(A[3], a2a, a2b);
        }

        // ---- D = sum_t mask_t(A) @ W[t]  (per-lane selects, no EXEC div) -
        v8f acc[N_TILES_OUT];
#pragma unroll
        for (int n = 0; n < N_TILES_OUT; ++n) acc[n] = (v8f){};
        const v16bf zeroV = {};
#pragma unroll
        for (int t = 0; t < N_TYPES; ++t) {
            bool m = (ety == t);
#pragma unroll
            for (int c = 0; c < N_KCHUNK; ++c) {
                v16bf am = m ? A[c] : zeroV;
#pragma unroll
                for (int n = 0; n < N_TILES_OUT; ++n) {
                    int f = (t * N_KCHUNK + c) * N_TILES_OUT + n;
                    v16bf B = *(const v16bf*)(sW + f * FRAG_ELEMS + lane * 16);
                    acc[n] = __builtin_amdgcn_wmma_f32_16x16x32_bf16(
                        false, am, false, B, (short)0, acc[n], false, false);
                }
            }
        }

        // ---- Segmented (dst-sorted) reduce across rows, then scatter -----
        // Row M = r + 8*h lives in acc[.][r]; rows 8..15 reach lanes 0..15
        // via shfl from lane col+16. Segment boundaries are wave-uniform.
        float run[N_TILES_OUT];
#pragma unroll
        for (int n = 0; n < N_TILES_OUT; ++n) run[n] = 0.0f;
        int curdst = -1;
#pragma unroll
        for (int M = 0; M < TILE_EDGES; ++M) {
            int dstM = __shfl(dst, M, 32);   // wave-uniform
            int tyM  = __shfl(ety, M, 32);   // wave-uniform
            if (dstM != curdst) {
                if (curdst >= 0 && h == 0) {
                    float* orow = out + (long)curdst * D_OUT;
#pragma unroll
                    for (int n = 0; n < N_TILES_OUT; ++n)
                        atomicAdd(orow + n * 16 + col, 0.25f * run[n]);
                }
#pragma unroll
                for (int n = 0; n < N_TILES_OUT; ++n) run[n] = 0.0f;
                curdst = dstM;
            }
            if (tyM >= 0) {
#pragma unroll
                for (int n = 0; n < N_TILES_OUT; ++n) {
                    float v = (M < 8) ? acc[n][M]
                                      : __shfl(acc[n][M - 8], col + 16, 32);
                    run[n] += v + sBias[tyM * D_OUT + n * 16 + col];
                }
            }
        }
        if (curdst >= 0 && h == 0) {
            float* orow = out + (long)curdst * D_OUT;
#pragma unroll
            for (int n = 0; n < N_TILES_OUT; ++n)
                atomicAdd(orow + n * 16 + col, 0.25f * run[n]);
        }
    }
}

// ---------------------------------------------------------------------------
extern "C" void kernel_launch(void* const* d_in, const int* in_sizes, int n_in,
                              void* d_out, int out_size, void* d_ws, size_t ws_size,
                              hipStream_t stream) {
    const float* x  = (const float*)d_in[0];
    const float* ef = (const float*)d_in[1];
    const int*   ei = (const int*)d_in[2];
    const int*   et = (const int*)d_in[3];
    const float* W  = (const float*)d_in[4];
    const float* b  = (const float*)d_in[5];
    float* out = (float*)d_out;

    // workspace layout (ints): bins[BINS] | bsum[1024 pad] | perm[N_EDGES]
    int* bins = (int*)d_ws;
    int* bsum = bins + BINS;
    int* perm = bins + BINS + 1024;

    // d_out is poisoned by the harness -> zero before atomic accumulation.
    int outN = N_NODES * D_OUT;
    zero_out_kernel<<<(outN + 255) / 256, 256, 0, stream>>>(out, outN);

    // counting sort of edges by key = dst*4 + type
    zero_bins_kernel<<<(BINS + 255) / 256, 256, 0, stream>>>(bins, BINS);
    hist_kernel<<<(N_EDGES + 255) / 256, 256, 0, stream>>>(ei, et, bins);
    scan_bins_kernel<<<NSB, 256, 0, stream>>>(bins, bsum);
    scan_sums_kernel<<<1, 1024, 0, stream>>>(bsum, NSB);
    add_offsets_kernel<<<(BINS + 255) / 256, 256, 0, stream>>>(bins, bsum);
    scatter_kernel<<<(N_EDGES + 255) / 256, 256, 0, stream>>>(ei, et, bins, perm);

    // fused gather -> bf16 WMMA -> segmented scatter
    int nTiles = (N_EDGES + TILE_EDGES - 1) / TILE_EDGES;  // 50000
    int blocks = 640;                                      // 5120 waves
    int totalWaves = blocks * WAVES_PER_BLOCK;
    rel_edge_conv_wmma<<<blocks, BLOCK_THREADS, SMEM_BYTES, stream>>>(
        x, ef, ei, et, W, b, perm, out, nTiles, totalWaves);
}
